// GraphAttention_66374424592902
// MI455X (gfx1250) — matrix operand order repaired and checked by reading, not compiled, not run on hardware
//
#include <hip/hip_runtime.h>
#include <hip/hip_bf16.h>

// GAT layer for MI455X (gfx1250, wave32).  B=8, N=2048, C=256.
// Phase 1: H = X @ W^T via v_wmma_f32_16x16x32_bf16 (f32 accum) + K-swizzled
//          bf16 copy of H so phase-3 B-fragments are contiguous 32B loads.
// Phase 2: s1/s2 = H @ a^T (wave-per-row shuffle reduction).
// Phase 3: flash-style masked softmax + P@H with bf16 WMMA; mask streamed once
//          (with global_prefetch); Hbf chunks staged block-wide into LDS via
//          async Global->LDS (ASYNCcnt), double-buffered.

typedef __attribute__((ext_vector_type(16))) __bf16 v16bf;
typedef __attribute__((ext_vector_type(8)))  float  v8f;
typedef __attribute__((ext_vector_type(4)))  int    v4i;

#define NB 8
#define NN 2048
#define NC 256
#define NEG_INF_F 1.0e12f
#define LEAKY_A 0.2f

#define __AS1 __attribute__((address_space(1)))
#define __AS3 __attribute__((address_space(3)))

#if defined(__has_builtin)
#  if __has_builtin(__builtin_amdgcn_global_load_async_to_lds_b128)
#    define HAS_ASYNC 1
#  endif
#endif
#ifndef HAS_ASYNC
#  define HAS_ASYNC 0
#endif

// Cooperative flat copy of one 16KB Hbf chunk (32 keys x 256 cols bf16) into
// LDS: 256 threads x 4 x b128. Async path uses the gfx1250 ASYNCcnt engine.
__device__ __forceinline__ void cp_chunk(const __bf16* __restrict__ g,
                                         __bf16* l, int tid)
{
#pragma unroll
    for (int q = 0; q < 4; ++q) {
        const int off = (tid + q * 256) * 8;    // 8 bf16 = 16B per b128
#if HAS_ASYNC
        __builtin_amdgcn_global_load_async_to_lds_b128(
            (__AS1 v4i*)(g + off), (__AS3 v4i*)(l + off), 0, 0);
#else
        *(float4*)(l + off) = *(const float4*)(g + off);
#endif
    }
}

template <int N>
__device__ __forceinline__ void wait_async()
{
#if HAS_ASYNC
#  if __has_builtin(__builtin_amdgcn_s_wait_asynccnt)
    __builtin_amdgcn_s_wait_asynccnt((unsigned short)N);
#  else
    asm volatile("s_wait_asynccnt %0" :: "n"(N) : "memory");
#  endif
#endif
}

// ---------------------------------------------------------------------------
// Phase 1: H[r, o] = sum_k X[r, k] * W[o, k]   (r = b*N + n flattened, 16384)
// Block: 256 thr = 8 waves. Wave (rg, cg): rows blockIdx*32 + rg*16, cols cg*64.
// Also writes Hbf swizzled: Hbf[((r/16)*256 + o)*16 + (r%16)].
// ---------------------------------------------------------------------------
__global__ __launch_bounds__(256) void
proj_kernel(const float* __restrict__ X, const float* __restrict__ W,
            float* __restrict__ H, __bf16* __restrict__ Hbf)
{
    const int lane = threadIdx.x & 31;
    const int wave = threadIdx.x >> 5;
    const int r    = lane & 15;        // A row / B,C col within tile
    const int half = lane >> 4;        // K-half select per ISA 16-bit layouts
    const int rowBase = blockIdx.x * 32 + (wave >> 2) * 16;
    const int colBase = (wave & 3) * 64;

    v8f acc[4] = {};
    const float* xrow = X + (size_t)(rowBase + r) * NC;

    for (int k0 = 0; k0 < NC; k0 += 32) {
        // A fragment (16x32 bf16): element e -> K = k0 + half*16 + e
        v16bf af;
#pragma unroll
        for (int q = 0; q < 4; ++q) {
            const int koff = k0 + half * 16 + q * 4;
            float4 v = *(const float4*)(xrow + koff);
            const int e0 = q * 4;
            af[e0 + 0] = (__bf16)v.x; af[e0 + 1] = (__bf16)v.y;
            af[e0 + 2] = (__bf16)v.z; af[e0 + 3] = (__bf16)v.w;
        }
#pragma unroll
        for (int t = 0; t < 4; ++t) {
            // B fragment (32x16 bf16): lane col = r, element e -> K = k0 + half*16 + e
            const int ocol = colBase + t * 16 + r;
            const float* wrow = W + (size_t)ocol * NC + k0 + half * 16;
            v16bf bf;
#pragma unroll
            for (int q = 0; q < 4; ++q) {
                float4 v = *(const float4*)(wrow + q * 4);
                bf[q * 4 + 0] = (__bf16)v.x; bf[q * 4 + 1] = (__bf16)v.y;
                bf[q * 4 + 2] = (__bf16)v.z; bf[q * 4 + 3] = (__bf16)v.w;
            }
            acc[t] = __builtin_amdgcn_wmma_f32_16x16x32_bf16(
                false, af, false, bf, (short)0, acc[t], false, false);
        }
    }

    // Epilogue: f32 row-major H + swizzled bf16 Hbf.
    const int m16 = rowBase >> 4; // global 16-row tile id (= b*128 + n/16)
#pragma unroll
    for (int t = 0; t < 4; ++t) {
        const int ocol = colBase + t * 16 + r;
#pragma unroll
        for (int i = 0; i < 8; ++i) {
            const int rw = i + half * 8;     // C/D layout: VGPR i -> row i + half*8
            const float v = acc[t][i];
            H[(size_t)(rowBase + rw) * NC + ocol] = v;
            Hbf[((size_t)m16 * NC + ocol) * 16 + rw] = (__bf16)v;
        }
    }
}

// ---------------------------------------------------------------------------
// Phase 2: s1[r] = H[r,:].a[0,:], s2[r] = H[r,:].a[1,:]. One wave per row.
// ---------------------------------------------------------------------------
__global__ __launch_bounds__(256) void
av_kernel(const float* __restrict__ H, const float* __restrict__ a,
          float* __restrict__ s1, float* __restrict__ s2)
{
    const int lane = threadIdx.x & 31;
    const int wave = threadIdx.x >> 5;
    const int row  = blockIdx.x * 8 + wave;
    const float* h = H + (size_t)row * NC + lane * 8;
    float d1 = 0.f, d2 = 0.f;
#pragma unroll
    for (int q = 0; q < 8; ++q) {
        const float hv = h[q];
        d1 += hv * a[lane * 8 + q];
        d2 += hv * a[NC + lane * 8 + q];
    }
#pragma unroll
    for (int off = 16; off; off >>= 1) {
        d1 += __shfl_xor(d1, off, 32);
        d2 += __shfl_xor(d2, off, 32);
    }
    if (lane == 0) { s1[row] = d1; s2[row] = d2; }
}

// ---------------------------------------------------------------------------
// Phase 3: out[b,n,:] = softmax_m(mask(leaky(s1[n]+s2[m]))) @ H[b,:,:] + bias.
// Wave owns 16 query rows x 256 cols (16 accum tiles). Online softmax; row
// state replicated in lane pair (r, r+16), combined via shfl_xor(16).
// All 8 waves share b and iterate m0 in lockstep, so each 32-key Hbf chunk
// (16KB, contiguous in the swizzled layout) is async-staged into LDS once per
// block, double-buffered: copy of chunk i+1 overlaps compute of chunk i.
// ---------------------------------------------------------------------------
__global__ __launch_bounds__(256) void
attn_kernel(const int* __restrict__ mask, const __bf16* __restrict__ Hbf,
            const float* __restrict__ s1, const float* __restrict__ s2,
            const float* __restrict__ bias, float* __restrict__ out)
{
    __shared__ __align__(32) __bf16 sB[2][32 * NC];   // 2 x 16KB

    const int tid  = threadIdx.x;
    const int lane = tid & 31;
    const int wave = tid >> 5;
    const int r    = lane & 15;
    const int half = lane >> 4;
    const int b    = blockIdx.x >> 4;              // 16 blocks per batch
    const int qRow = (blockIdx.x & 15) * 128 + wave * 16;

    const float  s1v  = s1[b * NN + qRow + r];
    const int*   mrow = mask + ((size_t)b * NN + qRow + r) * NN;
    const float* s2b  = s2 + b * NN;
    const __bf16* Hb  = Hbf + (size_t)b * NN * NC;  // chunk m0 base = Hb + m0*NC

    v8f   acc[16] = {};
    float Mx = -3.0e38f, S = 0.f;

    cp_chunk(Hb, sB[0], tid);                      // prologue: chunk 0 -> buf 0

    int cur = 0;
    for (int m0 = 0; m0 < NN; m0 += 32, cur ^= 1) {
        // issue async copy of next chunk into the other buffer, then wait for
        // the current chunk (loads complete in order: <=4 means chunk m0 done)
        if (m0 + 32 < NN) {
            cp_chunk(Hb + (size_t)(m0 + 32) * NC, sB[cur ^ 1], tid);
            wait_async<4>();
        } else {
            wait_async<0>();
        }
        __syncthreads();

        // keep this lane's mask row-stream ahead of the consumer
        if (m0 + 128 < NN) __builtin_prefetch(mrow + m0 + 128, 0, 1);

        // --- logits for this lane's 16 A-fragment elements ---
        float lg[16];
#pragma unroll
        for (int q = 0; q < 4; ++q) {
            const int koff = m0 + half * 16 + q * 4;
            const int4   mk = *(const int4*)(mrow + koff);
            const float4 sv = *(const float4*)(s2b + koff);
            const int e0 = q * 4;
            float l;
            l = s1v + sv.x; l = l > 0.f ? l : LEAKY_A * l; lg[e0+0] = mk.x > 0 ? l : l - NEG_INF_F;
            l = s1v + sv.y; l = l > 0.f ? l : LEAKY_A * l; lg[e0+1] = mk.y > 0 ? l : l - NEG_INF_F;
            l = s1v + sv.z; l = l > 0.f ? l : LEAKY_A * l; lg[e0+2] = mk.z > 0 ? l : l - NEG_INF_F;
            l = s1v + sv.w; l = l > 0.f ? l : LEAKY_A * l; lg[e0+3] = mk.w > 0 ? l : l - NEG_INF_F;
        }
        // --- online softmax update (state per row, replicated lane pair) ---
        float cmax = lg[0];
#pragma unroll
        for (int e = 1; e < 16; ++e) cmax = fmaxf(cmax, lg[e]);
        cmax = fmaxf(cmax, __shfl_xor(cmax, 16, 32));
        const float newM  = fmaxf(Mx, cmax);
        const float alpha = __expf(Mx - newM);
        float psum = 0.f;
        v16bf pa;
#pragma unroll
        for (int e = 0; e < 16; ++e) {
            const float p = __expf(lg[e] - newM);
            psum += p;
            pa[e] = (__bf16)p;
        }
        psum += __shfl_xor(psum, 16, 32);
        S  = S * alpha + psum;
        Mx = newM;

        // rescale accumulators: accum VGPR i holds row i + half*8
#pragma unroll
        for (int i = 0; i < 8; ++i) {
            const float af = __shfl(alpha, i + half * 8, 32);
#pragma unroll
            for (int t = 0; t < 16; ++t) acc[t][i] *= af;
        }

        // --- P @ Hbf : B-fragments are contiguous 32B reads from LDS ---
#pragma unroll
        for (int t = 0; t < 16; ++t) {
            const v16bf bf = *(const v16bf*)&sB[cur][(half * NC + t * 16 + r) * 16];
            acc[t] = __builtin_amdgcn_wmma_f32_16x16x32_bf16(
                false, pa, false, bf, (short)0, acc[t], false, false);
        }
        __syncthreads();   // all waves done with buf `cur` before it refills
    }

    // --- normalize, add bias, store row-major f32 ---
    float rec[8];
#pragma unroll
    for (int i = 0; i < 8; ++i) {
        const float Sr = __shfl(S, i + half * 8, 32);
        rec[i] = 1.0f / fmaxf(Sr, 1e-30f);
    }
#pragma unroll
    for (int t = 0; t < 16; ++t) {
        const int   ocol = t * 16 + r;
        const float bv   = bias[ocol];
#pragma unroll
        for (int i = 0; i < 8; ++i) {
            const int rw = i + half * 8;
            out[((size_t)b * NN + qRow + rw) * NC + ocol] = acc[t][i] * rec[i] + bv;
        }
    }
}

// ---------------------------------------------------------------------------
extern "C" void kernel_launch(void* const* d_in, const int* in_sizes, int n_in,
                              void* d_out, int out_size, void* d_ws, size_t ws_size,
                              hipStream_t stream)
{
    const float* x    = (const float*)d_in[0];
    const int*   mask = (const int*)  d_in[1];
    const float* W    = (const float*)d_in[2];
    const float* a    = (const float*)d_in[3];
    const float* bias = (const float*)d_in[4];
    float*       out  = (float*)d_out;

    char* ws = (char*)d_ws;
    float*  H   = (float*) ws;                                   // 16 MB
    __bf16* Hbf = (__bf16*)(ws + (size_t)16 * 1024 * 1024);      //  8 MB
    float*  s1  = (float*) (ws + (size_t)24 * 1024 * 1024);      // 64 KB
    float*  s2  = s1 + NB * NN;                                  // 64 KB

    proj_kernel<<<dim3((NB * NN) / 32), dim3(256), 0, stream>>>(x, W, H, Hbf);
    av_kernel  <<<dim3((NB * NN) / 8),  dim3(256), 0, stream>>>(H, a, s1, s2);
    attn_kernel<<<dim3(NB * 16),        dim3(256), 0, stream>>>(mask, Hbf, s1, s2, bias, out);
}
